// ExpDecayRNN_76716705841423
// MI455X (gfx1250) — matrix operand order repaired and checked by reading, not compiled, not run on hardware
//
#include <hip/hip_runtime.h>
#include <hip/hip_bf16.h>
#include <math.h>

// ---- shapes (compile-time for this problem) ----
#define BB 64
#define TT 1024
#define DD 128
#define HH 256
#define NG 1024            // combined gate columns: r(256)+z(256)+inn(256)+hn(256)
#define KK 384             // combined K: D(128) + H(256)
#define NTILES (NG/16)     // 64
#define KSTEPS (KK/32)     // 12

typedef __attribute__((ext_vector_type(16))) _Float16 v16h;
typedef __attribute__((ext_vector_type(8)))  _Float16 v8h;
typedef __attribute__((ext_vector_type(8)))  float    v8f;

// load a v16h from two 16-byte LDS/global chunks
__device__ __forceinline__ v16h ld2(const _Float16* p0, const _Float16* p1) {
    v8h lo = *(const v8h*)p0;
    v8h hi = *(const v8h*)p1;
    return __builtin_shufflevector(lo, hi, 0,1,2,3,4,5,6,7,8,9,10,11,12,13,14,15);
}

__device__ __forceinline__ v8f wmma_f16(v16h a, v16h b, v8f c) {
    return __builtin_amdgcn_wmma_f32_16x16x32_f16(
        /*neg_a=*/false, a, /*neg_b=*/false, b,
        /*c_mod=*/(short)0, c, /*reuse_a=*/false, /*reuse_b=*/false);
}

__device__ __forceinline__ float sigmoidf_(float x) { return 1.0f / (1.0f + expf(-x)); }
__device__ __forceinline__ float reluf_(float x)    { return fmaxf(x, 0.0f); }

// low 32 bits of a flat pointer to LDS == LDS byte offset (ISA aperture rule)
__device__ __forceinline__ unsigned int lds_off(const void* p) {
    return (unsigned int)(uintptr_t)p;
}

// =====================================================================
// Kernel 1: per-(b,d) forward-fill scan. 8192 independent sequences.
// =====================================================================
__global__ __launch_bounds__(256) void scan_kernel(
    const float* __restrict__ x, const unsigned char* __restrict__ mask,
    const float* __restrict__ ts,
    float* __restrict__ ffill, float* __restrict__ idelta, float* __restrict__ meanv)
{
    int g = blockIdx.x * 256 + threadIdx.x;       // 0..8191
    int b = g >> 7;
    int d = g & 127;
    const size_t base = (size_t)b * TT * DD + d;
    const float* tsb = ts + (size_t)b * TT;

    float px = 0.0f, ptd = 0.0f, sum = 0.0f, cnt = 0.0f, prev_ts = 0.0f;
    for (int t = 0; t < TT; ++t) {
        size_t ix = base + (size_t)t * DD;
        float xt = x[ix];
        bool  m  = mask[ix] != 0;                 // True == missing
        float tc = tsb[t];
        float dt = (t == 0) ? 1.0f : (tc - prev_ts);
        prev_ts = tc;
        if (!m) px = xt;
        ptd += dt;
        ffill[ix] = px;
        idelta[ix] = fminf(fmaxf(ptd - 1.0f, 0.0f), 1000.0f);
        if (m) cnt += 1.0f; else sum += xt;
        if (!m) ptd = 0.0f;
    }
    meanv[b * DD + d] = sum / fmaxf(cnt, 1.0f);
}

// =====================================================================
// Kernel 2: combined-weight prep, pre-swizzled to the per-lane WMMA
// B-fragment layout so the RNN kernel does one contiguous 32B load/lane.
// =====================================================================
__device__ __forceinline__ float Wcomb(int n, int k,
    const float* w_ih, const float* w_hh)
{
    if (n < 512)      return (k < DD) ? w_ih[n * DD + k] : w_hh[n * HH + (k - DD)];
    else if (n < 768) return (k < DD) ? w_ih[n * DD + k] : 0.0f;                   // inn: x-only
    else              return (k < DD) ? 0.0f : w_hh[(n - 256) * HH + (k - DD)];    // hn:  h-only
}

__global__ __launch_bounds__(256) void wprep_kernel(
    const float* __restrict__ w_ih, const float* __restrict__ w_hh,
    const float* __restrict__ b_ih, const float* __restrict__ b_hh,
    _Float16* __restrict__ wt, float* __restrict__ biasc)
{
    int g = blockIdx.x * 256 + threadIdx.x;
    if (g < NTILES * KSTEPS * 32) {
        int nt   = g / (KSTEPS * 32);
        int rem  = g % (KSTEPS * 32);
        int ks   = rem / 32;
        int lane = rem % 32;
        int n  = nt * 16 + (lane & 15);
        int kh = lane >> 4;
        _Float16* o = wt + (size_t)g * 16;
        #pragma unroll
        for (int j = 0; j < 16; ++j) {
            int k = ks * 32 + kh * 16 + j;
            o[j] = (_Float16)Wcomb(n, k, w_ih, w_hh);
        }
    } else if (g < NTILES * KSTEPS * 32 + NG) {
        int n = g - NTILES * KSTEPS * 32;
        float v;
        if (n < 512)      v = b_ih[n] + b_hh[n];
        else if (n < 768) v = b_ih[n];            // inn bias
        else              v = b_hh[n - 256];      // hn bias
        biasc[n] = v;
    }
}

// =====================================================================
// Kernel 3: fill_w GEMM (WMMA) + compose final input as f16 [T][B][D].
// =====================================================================
__global__ __launch_bounds__(256) void fill_kernel(
    const float* __restrict__ idelta, const float* __restrict__ ffill,
    const float* __restrict__ x, const unsigned char* __restrict__ mask,
    const float* __restrict__ meanv,
    const float* __restrict__ idw, const float* __restrict__ idb,
    _Float16* __restrict__ xh)
{
    __shared__ _Float16 idw_s[DD][136];   // padded rows: 272B stride, 16B aligned
    __shared__ _Float16 ast[64][136];

    int tid = threadIdx.x;
    int rowbase = blockIdx.x * 64;

    for (int i = 0; i < 64; ++i) {
        int idx = tid + i * 256;              // 16384 elems
        int j = idx >> 7, k = idx & 127;
        idw_s[j][k] = (_Float16)idw[j * DD + k];
    }
    for (int i = 0; i < 32; ++i) {
        int idx = tid + i * 256;              // 8192 elems
        int r = idx >> 7, k = idx & 127;
        ast[r][k] = (_Float16)idelta[(size_t)(rowbase + r) * DD + k];
    }
    __syncthreads();

    int wave = tid >> 5, lane = tid & 31;
    int lrow = lane & 15, kh = lane >> 4;
    int mt  = wave & 3;                        // 4 M-tiles of 16 rows
    int ntb = (wave >> 2) * 4;                 // 8 N-tiles over 2 wave groups

    for (int nt = ntb; nt < ntb + 4; ++nt) {
        v8f c = {};
        #pragma unroll
        for (int ks = 0; ks < 4; ++ks) {
            const _Float16* ap = &ast[mt * 16 + lrow][ks * 32 + 8 * kh];
            v16h a = ld2(ap, ap + 16);                      // A: interleaved 8-runs
            const _Float16* bp = &idw_s[nt * 16 + lrow][ks * 32 + 8 * kh];
            v16h bmat = ld2(bp, bp + 16);                   // B: interleaved 8-runs
            c = wmma_f16(a, bmat, c);
        }
        int col  = nt * 16 + lrow;
        float bias = idb[col];
        #pragma unroll
        for (int v = 0; v < 8; ++v) {
            int row = rowbase + mt * 16 + v + 8 * kh;
            int bi = row >> 10, t = row & 1023;             // row = b*T + t
            float fw = expf(-reluf_(c[v] + bias));
            size_t ix = (size_t)row * DD + col;
            float fl = ffill[ix] * fw + (1.0f - fw) * meanv[bi * DD + col];
            float xv = mask[ix] ? fl : x[ix];
            xh[((size_t)t * BB + bi) * DD + col] = (_Float16)xv;
        }
    }
}

// =====================================================================
// Kernel 4: sequential recurrence. 4 persistent WGs (one per 16-row
// batch group), LDS barriers only. x_t tiles are double-buffered into
// LDS via global_load_async_to_lds_b64 (ASYNCcnt), issued one step
// ahead and overlapped with the WMMA weight-stream GEMM.
// =====================================================================
__global__ __launch_bounds__(1024, 1) void rnn_kernel(
    const _Float16* __restrict__ xh, const _Float16* __restrict__ wt,
    const float* __restrict__ biasc, const float* __restrict__ ts,
    const float* __restrict__ hdw, const float* __restrict__ hdb,
    float* __restrict__ out)
{
    __shared__ float    h_s[16][HH];          // 16 KB, f32 master copy of h
    __shared__ _Float16 hst[16][264];         // h_dec f16 staging (528B stride = 33*16)
    __shared__ _Float16 xbuf[2][16][136];     // async x tiles, double buffered
    __shared__ float    gbuf[16][NG + 4];     // 64.25 KB, padded

    int tid = threadIdx.x;
    int bbase = blockIdx.x * 16;
    int wave = tid >> 5, lane = tid & 31;
    int lrow = lane & 15, kh = lane >> 4;
    int nt0 = 2 * wave, nt1 = 2 * wave + 1;

    #pragma unroll
    for (int i = 0; i < 4; ++i) {
        int idx = tid + i * 1024;
        h_s[idx >> 8][idx & 255] = 0.0f;
    }

    // issue async load of x tile for t=0 into buffer 0 (8B per lane, waves 0-15)
    int xm = tid >> 5;           // row 0..15 (valid when tid<512)
    int xd = (tid & 31) * 4;     // 4 f16 per lane
    if (tid < 512) {
        unsigned int       ldst = lds_off(&xbuf[0][xm][xd]);
        unsigned long long gsrc =
            (unsigned long long)(uintptr_t)(xh + ((size_t)bbase + xm) * DD + xd);
        asm volatile("global_load_async_to_lds_b64 %0, %1, off"
                     :: "v"(ldst), "v"(gsrc) : "memory");
    }
    asm volatile("s_wait_asynccnt 0x0" ::: "memory");
    __syncthreads();

    for (int t = 0; t < TT; ++t) {
        // ---- phase 1: decay h -> f16 staging; prefetch next x tile ----
        #pragma unroll
        for (int i = 0; i < 4; ++i) {
            int idx = tid + i * 1024;
            int m = idx >> 8, c = idx & 255;
            int b = bbase + m;
            float tc = ts[(size_t)b * TT + t];
            float dt = (t == 0) ? 1.0f : (tc - ts[(size_t)b * TT + t - 1]);
            float tdl = logf(fminf(dt, 1000.0f));
            float dec = expf(-reluf_(tdl * hdw[c] + hdb[c]));
            float h = h_s[m][c] * dec;
            h_s[m][c] = h;
            hst[m][c] = (_Float16)h;
        }
        if (t + 1 < TT) {
            if (tid < 512) {
                unsigned int       ldst = lds_off(&xbuf[(t + 1) & 1][xm][xd]);
                unsigned long long gsrc = (unsigned long long)(uintptr_t)
                    (xh + ((size_t)(t + 1) * BB + bbase + xm) * DD + xd);
                asm volatile("global_load_async_to_lds_b64 %0, %1, off"
                             :: "v"(ldst), "v"(gsrc) : "memory");
            }
            asm volatile("s_wait_asynccnt 0x1" ::: "memory");  // tile t done, t+1 in flight
        } else {
            asm volatile("s_wait_asynccnt 0x0" ::: "memory");
        }
        __syncthreads();

        // ---- phase 2: G = [x_t | h_dec] @ Wcomb^T ; weights streamed from L2 ----
        const _Float16 (* __restrict__ xcur)[136] = xbuf[t & 1];
        v8f c0 = {}, c1 = {};
        const _Float16* wbase0 = wt + ((size_t)nt0 * KSTEPS) * 512 + (size_t)lane * 16;
        const _Float16* wbase1 = wt + ((size_t)nt1 * KSTEPS) * 512 + (size_t)lane * 16;
        __builtin_prefetch(wbase0, 0, 1);
        __builtin_prefetch(wbase1, 0, 1);
        #pragma unroll
        for (int ks = 0; ks < KSTEPS; ++ks) {
            const _Float16* ap = (ks < 4)
                ? &xcur[lrow][ks * 32 + 16 * kh]
                : &hst[lrow][(ks - 4) * 32 + 16 * kh];
            v16h a = ld2(ap, ap + 8);
            v16h b0 = *(const v16h*)(wbase0 + (size_t)ks * 512);
            c0 = wmma_f16(a, b0, c0);
            v16h b1 = *(const v16h*)(wbase1 + (size_t)ks * 512);
            c1 = wmma_f16(a, b1, c1);
        }
        int mrow = 8 * kh;
        int n0 = nt0 * 16 + lrow, n1 = nt1 * 16 + lrow;
        #pragma unroll
        for (int v = 0; v < 8; ++v) {
            gbuf[mrow + v][n0] = c0[v];
            gbuf[mrow + v][n1] = c1[v];
        }
        __syncthreads();

        // ---- phase 3: gates + state update + output ----
        #pragma unroll
        for (int i = 0; i < 4; ++i) {
            int idx = tid + i * 1024;
            int m = idx >> 8, c = idx & 255;
            int b = bbase + m;
            float r   = sigmoidf_(gbuf[m][c]        + biasc[c]);
            float z   = sigmoidf_(gbuf[m][256 + c]  + biasc[256 + c]);
            float inn =          gbuf[m][512 + c]   + biasc[512 + c];
            float hn  =          gbuf[m][768 + c]   + biasc[768 + c];
            float n   = tanhf(inn + r * hn);
            float hprev = h_s[m][c];                // decayed h
            float hnew  = (1.0f - z) * n + z * hprev;
            h_s[m][c] = hnew;
            out[((size_t)b * TT + t) * HH + c] = hnew;
        }
        __syncthreads();
    }

    // h_last appended after hidden
    #pragma unroll
    for (int i = 0; i < 4; ++i) {
        int idx = tid + i * 1024;
        int m = idx >> 8, c = idx & 255;
        out[(size_t)BB * TT * HH + (size_t)(bbase + m) * HH + c] = h_s[m][c];
    }
}

// =====================================================================
extern "C" void kernel_launch(void* const* d_in, const int* in_sizes, int n_in,
                              void* d_out, int out_size, void* d_ws, size_t ws_size,
                              hipStream_t stream) {
    const float*         x    = (const float*)d_in[0];
    const unsigned char* mask = (const unsigned char*)d_in[1];
    const float*         ts   = (const float*)d_in[2];
    // d_in[3] = init_time: unused (time_delta[:,0] == 1.0 analytically)
    const float* w_ih = (const float*)d_in[4];
    const float* w_hh = (const float*)d_in[5];
    const float* b_ih = (const float*)d_in[6];
    const float* b_hh = (const float*)d_in[7];
    const float* idw  = (const float*)d_in[8];
    const float* idb  = (const float*)d_in[9];
    const float* hdw  = (const float*)d_in[10];   // (H,1) -> flat H
    const float* hdb  = (const float*)d_in[11];
    float* out = (float*)d_out;

    // workspace carve-up (all offsets 32B aligned)
    char* ws = (char*)d_ws;
    float*    ffill  = (float*)(ws);                       // 33554432 B
    float*    idelta = (float*)(ws + 33554432u);           // 33554432 B
    float*    meanv  = (float*)(ws + 67108864u);           //    32768 B
    _Float16* xh     = (_Float16*)(ws + 67141632u);        // 16777216 B
    _Float16* wt     = (_Float16*)(ws + 83918848u);        //   786432 B
    float*    biasc  = (float*)(ws + 84705280u);           //     4096 B

    scan_kernel<<<32, 256, 0, stream>>>(x, mask, ts, ffill, idelta, meanv);
    wprep_kernel<<<100, 256, 0, stream>>>(w_ih, w_hh, b_ih, b_hh, wt, biasc);
    fill_kernel<<<1024, 256, 0, stream>>>(idelta, ffill, x, mask, meanv, idw, idb, xh);
    rnn_kernel<<<4, 1024, 0, stream>>>(xh, wt, biasc, ts, hdw, hdb, out);
}
